// DecayAttention_23330262352167
// MI455X (gfx1250) — matrix operand order repaired and checked by reading, not compiled, not run on hardware
//
#include <hip/hip_runtime.h>

// DecayAttention for MI455X (gfx1250, wave32, WMMA bf16 16x16x32, fp32 accum)
// LDS-tiled GEMMs + flash attention, async global->LDS staging (ASYNCcnt),
// ping-pong double buffering. B=2, T=2048, C=1024, H=16, DH=64

#define B_  2
#define T_  2048
#define C_  1024
#define H_  16
#define DH_ 64

typedef __bf16 bf16t;
typedef bf16t v16bf __attribute__((ext_vector_type(16)));
typedef float v8f   __attribute__((ext_vector_type(8)));
typedef int   v4i_vs __attribute__((vector_size(16)));   // matches async-LDS builtin param

#if defined(__HIP_DEVICE_COMPILE__) && __has_builtin(__builtin_amdgcn_global_load_async_to_lds_b128)
#define HAS_ASYNC_LDS 1
#else
#define HAS_ASYNC_LDS 0
#endif

#define BF16_WMMA(a, b, c) \
  __builtin_amdgcn_wmma_f32_16x16x32_bf16(false, (a), false, (b), (short)0, (c), false, false)

// ---------- helpers ----------

static __device__ __forceinline__ unsigned short f2bf(float f) {
  unsigned int u = __float_as_uint(f);
  u += 0x7FFFu + ((u >> 16) & 1u);   // round-to-nearest-even
  return (unsigned short)(u >> 16);
}

static __device__ __forceinline__ v16bf ld_frag2(const unsigned short* lo,
                                                 const unsigned short* hi) {
  union { v16bf v; uint4 q[2]; } u;
  u.q[0] = *reinterpret_cast<const uint4*>(lo);
  u.q[1] = *reinterpret_cast<const uint4*>(hi);
  return u.v;
}

static __device__ __forceinline__ v16bf ld_frag32(const unsigned short* p) {
  return ld_frag2(p, p + 8);
}

// 16-byte global -> LDS copy; async (ASYNCcnt-tracked) path when available.
static __device__ __forceinline__ void async_cp16(const unsigned short* g,
                                                  unsigned short* l) {
#if HAS_ASYNC_LDS
  __builtin_amdgcn_global_load_async_to_lds_b128((v4i_vs*)g, (v4i_vs*)l, 0, 0);
#else
  *reinterpret_cast<uint4*>(l) = *reinterpret_cast<const uint4*>(g);
#endif
}

static __device__ __forceinline__ void wait_stage() {
#if HAS_ASYNC_LDS
#if __has_builtin(__builtin_amdgcn_s_wait_asynccnt)
  __builtin_amdgcn_s_wait_asynccnt(0);
#else
  asm volatile("s_wait_asynccnt 0x0" ::: "memory");
#endif
#endif
  __syncthreads();
}

// ---------- fp32 -> bf16 converter ----------

__global__ __launch_bounds__(256) void k_cvt_bf16(const float* __restrict__ src,
                                                  unsigned short* __restrict__ dst,
                                                  int n) {
  int i = blockIdx.x * 256 + threadIdx.x;
  if (i < n) dst[i] = f2bf(src[i]);
}

// ---------- shared GEMM mainloop ----------
// WG = 256 threads = 8 waves as 2(M) x 4(N). Block tile 64(M) x 256(N), K-step 32.
// A [M,K], B [N,K] row-major bf16. LDS padded stride 40 halfs (20 dwords):
// (20*lane)%64 distinct for all 16 lanes -> conflict-free ds_load_b128.

#define GA_STR 40
#define GB_STR 40
#define GA_BUF (64 * GA_STR)
#define GB_BUF (256 * GB_STR)

static __device__ __forceinline__ void gemm_mainloop(
    const unsigned short* __restrict__ Ab, const unsigned short* __restrict__ Bb,
    int m0, int n0, int K, unsigned short* sA, unsigned short* sB,
    v8f (&acc)[2][4]) {
  const int tid = threadIdx.x;
  const int lane = tid & 31;
  const int wave = tid >> 5;
  const int li = lane & 15, hi = lane >> 4;
  const int mr = wave >> 2, nc = wave & 3;
  const int selA = hi * 8, selB = hi * 8;

  const int arow = tid >> 2, ac8 = (tid & 3) * 8;   // A: 64 rows x 4 chunks

  auto stage = [&](int k0, int buf) {
    async_cp16(Ab + (size_t)(m0 + arow) * K + k0 + ac8,
               sA + buf * GA_BUF + arow * GA_STR + ac8);
#pragma unroll
    for (int it = 0; it < 4; it++) {                // B: 256 rows x 4 chunks
      const int chunk = tid + 256 * it;
      const int row = chunk >> 2, c8 = (chunk & 3) * 8;
      async_cp16(Bb + (size_t)(n0 + row) * K + k0 + c8,
                 sB + buf * GB_BUF + row * GB_STR + c8);
    }
  };

  stage(0, 0);
  wait_stage();

  const int NS = K / 32;
  // unroll 1: keeps the 8 accumulator tuples pinned across iterations
  // (unroll-by-2 made the RA ping-pong acc tuples -> v_mov + hazard-nop storms)
#pragma unroll 1
  for (int s = 0; s < NS; s++) {
    const int cur = s & 1;
    if (s + 1 < NS) stage((s + 1) * 32, cur ^ 1);

    v16bf afr[2];
#pragma unroll
    for (int r = 0; r < 2; r++) {
      const unsigned short* p =
          sA + cur * GA_BUF + (mr * 32 + r * 16 + li) * GA_STR + selA;
      afr[r] = ld_frag2(p, p + 16);
    }
#pragma unroll
    for (int j = 0; j < 4; j++) {
      const unsigned short* p =
          sB + cur * GB_BUF + (nc * 64 + j * 16 + li) * GB_STR + selB;
      v16bf bfr = ld_frag2(p, p + 16);
      acc[0][j] = BF16_WMMA(afr[0], bfr, acc[0][j]);
      acc[1][j] = BF16_WMMA(afr[1], bfr, acc[1][j]);
    }
    wait_stage();
  }
}

// ---------- GEMM 1: qkv = x @ qkv_w^T + qkv_b, scatter to Q/K/Vt ----------

__global__ __launch_bounds__(256) void k_gemm_qkv(
    const unsigned short* __restrict__ xb,   // [4096,1024]
    const unsigned short* __restrict__ wb,   // [3072,1024]
    const float* __restrict__ bias,          // [3072]
    unsigned short* __restrict__ qbuf,       // [B,H,T,DH]
    unsigned short* __restrict__ kbuf,       // [B,H,T,DH]
    unsigned short* __restrict__ vtbuf) {    // [B,H,DH,T]
  __shared__ unsigned short sA[2 * GA_BUF];
  __shared__ unsigned short sB[2 * GB_BUF];

  const int lane = threadIdx.x & 31;
  const int wave = threadIdx.x >> 5;
  const int li = lane & 15, hi = lane >> 4;
  const int mr = wave >> 2, nc = wave & 3;
  const int NT = (3 * C_) / 256;                     // 12
  const int m0 = (blockIdx.x / NT) * 64;
  const int n0 = (blockIdx.x % NT) * 256;

  v8f acc[2][4];
#pragma unroll
  for (int r = 0; r < 2; r++)
#pragma unroll
    for (int j = 0; j < 4; j++) acc[r][j] = {};

  gemm_mainloop(xb, wb, m0, n0, C_, sA, sB, acc);

#pragma unroll
  for (int r = 0; r < 2; r++) {
#pragma unroll
    for (int j = 0; j < 4; j++) {
      const int o = n0 + nc * 64 + j * 16 + li;   // [0, 3C)
      const int which = o >> 10;
      const int rem = o & (C_ - 1);
      const int h = rem >> 6;
      const int d = rem & (DH_ - 1);
      const float bv = bias[o];
#pragma unroll
      for (int e = 0; e < 8; e++) {
        const int m = m0 + mr * 32 + r * 16 + e + 8 * hi;
        const int bb = m >> 11;
        const int t = m & (T_ - 1);
        const unsigned short val = f2bf(acc[r][j][e] + bv);
        if (which == 0)
          qbuf[((size_t)(bb * H_ + h) * T_ + t) * DH_ + d] = val;
        else if (which == 1)
          kbuf[((size_t)(bb * H_ + h) * T_ + t) * DH_ + d] = val;
        else
          vtbuf[((size_t)(bb * H_ + h) * DH_ + d) * T_ + t] = val;
      }
    }
  }
}

// ---------- Flash attention with multiplicative decay mask ----------
// One wave = 16 query rows; 8 waves/WG = 128 rows. K/Vt blocks (32 keys)
// staged in LDS via async copies, ping-pong buffered. Computes S^T = K*Q^T so
// the f32 D-fragment feeds straight into the P*V A-fragment (no shuffles).

#define SK_STR 72                 // 64 data halfs + pad (36 dwords: conflict-free)
#define SV_STR 40
#define SK_BUF (32 * SK_STR)
#define SV_BUF (64 * SV_STR)

__global__ __launch_bounds__(256) void k_attn(
    const unsigned short* __restrict__ qbuf,   // [B,H,T,DH]
    const unsigned short* __restrict__ kbuf,   // [B,H,T,DH]
    const unsigned short* __restrict__ vtbuf,  // [B,H,DH,T]
    const float* __restrict__ lambda_raw,      // [H]
    unsigned short* __restrict__ obuf) {       // [B,T,C]
  __shared__ unsigned short sK[2 * SK_BUF];
  __shared__ unsigned short sV[2 * SV_BUF];

  const int tid = threadIdx.x;
  const int lane = tid & 31;
  const int wave = tid >> 5;
  const int li = lane & 15, hi = lane >> 4;

  const int IB = T_ / 128;                  // 16 row-blocks per (b,h)
  const int bh = blockIdx.x / IB;
  const int ib = blockIdx.x % IB;
  const int bb = bh / H_, h = bh % H_;
  const int ibase = ib * 128 + wave * 16;

  const size_t hoff = (size_t)bh * T_ * DH_;
  const unsigned short* Q = qbuf + hoff;
  const unsigned short* K = kbuf + hoff;
  const unsigned short* Vt = vtbuf + hoff;   // [DH][T]

  const float lam = logf(1.0f + __expf(lambda_raw[h]));  // softplus
  const float scale = 0.125f;                            // DH^-0.5

  // Stage one 32-key block: K rows (32x32, d 0..31) + Vt rows (64x16, j 0..15). 1 chunk/thread.
  auto stage = [&](int j0, int buf) {
    if (tid < 128) {
      const int row = tid >> 2, c8 = (tid & 3) * 8;
      async_cp16(K + (size_t)(j0 + row) * DH_ + c8,
                 sK + buf * SK_BUF + row * SK_STR + c8);
    } else {
      const int t2 = tid - 128;
      const int row = t2 >> 1, c8 = (t2 & 1) * 8;
      async_cp16(Vt + (size_t)row * T_ + j0 + c8,
                 sV + buf * SV_BUF + row * SV_STR + c8);
    }
  };

  // Q as B-fragments of S^T (lane holds query row ibase+li, d in 8-wide halves split by lane half).
  const unsigned short* qp = Q + (size_t)(ibase + li) * DH_ + hi * 8;
  const v16bf q0 = ld_frag2(qp, qp + 16);
  const v16bf q1 = ld_frag2(qp + 32, qp + 48);

  v8f oacc[4];
#pragma unroll
  for (int dt = 0; dt < 4; dt++) oacc[dt] = {};
  float m_run = -1e30f, l_run = 0.0f;
  const float iF = (float)(ibase + li);

  stage(0, 0);
  wait_stage();

#pragma unroll 1
  for (int j0 = 0; j0 < T_; j0 += 32) {
    const int cur = (j0 >> 5) & 1;
    if (j0 + 32 < T_) stage(j0 + 32, cur ^ 1);

    // ---- S^T tiles (rows j, cols i) from LDS ----
    v8f st[2];
#pragma unroll
    for (int t = 0; t < 2; t++) {
      const unsigned short* pk =
          sK + cur * SK_BUF + (t * 16 + li) * SK_STR + hi * 8;
      v16bf ka0 = ld_frag2(pk, pk + 16);        // d = 0..31
      v16bf ka1 = ld_frag2(pk + 32, pk + 48);   // d = 32..63
      v8f z = {};
      z = BF16_WMMA(ka0, q0, z);
      st[t] = BF16_WMMA(ka1, q1, z);
    }

    // ---- logits with multiplicative decay; online softmax ----
    float lg[2][8];
    float lmax = -1e30f;
#pragma unroll
    for (int t = 0; t < 2; t++) {
#pragma unroll
      for (int e = 0; e < 8; e++) {
        const float jF = (float)(j0 + t * 16 + e + 8 * hi);
        const float dec = __expf(-lam * fabsf(iF - jF));
        const float v = st[t][e] * scale * dec;
        lg[t][e] = v;
        lmax = fmaxf(lmax, v);
      }
    }
    lmax = fmaxf(lmax, __shfl_xor(lmax, 16, 32));
    const float m_new = fmaxf(m_run, lmax);
    const float alpha = __expf(m_run - m_new);

    union { v16bf v; unsigned short s[16]; } pfrag;
    float ssum = 0.0f;
#pragma unroll
    for (int t = 0; t < 2; t++) {
#pragma unroll
      for (int e = 0; e < 8; e++) {
        const float pv = __expf(lg[t][e] - m_new);
        ssum += pv;
        pfrag.s[t * 8 + e] = f2bf(pv);
      }
    }
    ssum += __shfl_xor(ssum, 16, 32);
    l_run = l_run * alpha + ssum;
    m_run = m_new;

#pragma unroll
    for (int e = 0; e < 8; e++) {
      const float af = __shfl(alpha, e + 8 * hi, 32);
#pragma unroll
      for (int dt = 0; dt < 4; dt++) oacc[dt][e] *= af;
    }

    // ---- O += P * V from LDS (Vt rows contiguous in j) ----
#pragma unroll
    for (int dt = 0; dt < 4; dt++) {
      const unsigned short* pv =
          sV + cur * SV_BUF + (dt * 16 + li) * SV_STR + hi * 8;
      v16bf vb = ld_frag2(pv, pv + 16);
      oacc[dt] = BF16_WMMA(pfrag.v, vb, oacc[dt]);
    }

    wait_stage();
  }

  // ---- normalize and store [B,T,C] bf16 ----
#pragma unroll
  for (int e = 0; e < 8; e++) {
    const float lr = __shfl(l_run, e + 8 * hi, 32);
    const float inv = 1.0f / lr;
    const int irow = ibase + e + 8 * hi;
#pragma unroll
    for (int dt = 0; dt < 4; dt++) {
      obuf[((size_t)(bb * T_ + irow)) * C_ + h * DH_ + dt * 16 + li] =
          f2bf(oacc[dt][e] * inv);
    }
  }
}

// ---------- GEMM 2: out = attn_out @ proj_w^T + proj_b (fp32 out) ----------

__global__ __launch_bounds__(256) void k_gemm_proj(
    const unsigned short* __restrict__ ab,   // [4096,1024] bf16
    const unsigned short* __restrict__ wb,   // [1024,1024] bf16
    const float* __restrict__ bias,          // [1024]
    float* __restrict__ out) {               // [4096,1024] fp32
  __shared__ unsigned short sA[2 * GA_BUF];
  __shared__ unsigned short sB[2 * GB_BUF];

  const int lane = threadIdx.x & 31;
  const int wave = threadIdx.x >> 5;
  const int li = lane & 15, hi = lane >> 4;
  const int mr = wave >> 2, nc = wave & 3;
  const int NT = C_ / 256;                           // 4
  const int m0 = (blockIdx.x / NT) * 64;
  const int n0 = (blockIdx.x % NT) * 256;

  v8f acc[2][4];
#pragma unroll
  for (int r = 0; r < 2; r++)
#pragma unroll
    for (int j = 0; j < 4; j++) acc[r][j] = {};

  gemm_mainloop(ab, wb, m0, n0, C_, sA, sB, acc);

#pragma unroll
  for (int r = 0; r < 2; r++) {
#pragma unroll
    for (int j = 0; j < 4; j++) {
      const int o = n0 + nc * 64 + j * 16 + li;
      const float bv = bias[o];
#pragma unroll
      for (int e = 0; e < 8; e++) {
        const int m = m0 + mr * 32 + r * 16 + e + 8 * hi;
        out[(size_t)m * C_ + o] = acc[r][j][e] + bv;
      }
    }
  }
}

// ---------- launcher ----------

extern "C" void kernel_launch(void* const* d_in, const int* in_sizes, int n_in,
                              void* d_out, int out_size, void* d_ws, size_t ws_size,
                              hipStream_t stream) {
  (void)in_sizes; (void)n_in; (void)out_size; (void)ws_size;
  const float* x          = (const float*)d_in[0];
  const float* qkv_w      = (const float*)d_in[1];
  const float* qkv_b      = (const float*)d_in[2];
  const float* proj_w     = (const float*)d_in[3];
  const float* proj_b     = (const float*)d_in[4];
  const float* lambda_raw = (const float*)d_in[5];

  const size_t nX  = (size_t)B_ * T_ * C_;        // 4,194,304
  const size_t nWq = (size_t)3 * C_ * C_;         // 3,145,728
  const size_t nWp = (size_t)C_ * C_;             // 1,048,576
  const size_t nQ  = (size_t)B_ * H_ * T_ * DH_;  // 4,194,304

  char* ws = (char*)d_ws;
  size_t off = 0;
  auto alloc = [&](size_t elems) {
    unsigned short* p = (unsigned short*)(ws + off);
    off += ((elems * sizeof(unsigned short)) + 255) & ~(size_t)255;
    return p;
  };
  unsigned short* xb  = alloc(nX);
  unsigned short* wqb = alloc(nWq);
  unsigned short* wpb = alloc(nWp);
  unsigned short* qb  = alloc(nQ);
  unsigned short* kb  = alloc(nQ);
  unsigned short* vtb = alloc(nQ);
  unsigned short* ob  = alloc(nX);

  k_cvt_bf16<<<(int)((nX  + 255) / 256), 256, 0, stream>>>(x,      xb,  (int)nX);
  k_cvt_bf16<<<(int)((nWq + 255) / 256), 256, 0, stream>>>(qkv_w,  wqb, (int)nWq);
  k_cvt_bf16<<<(int)((nWp + 255) / 256), 256, 0, stream>>>(proj_w, wpb, (int)nWp);

  // QKV projection: (4096/64) * (3072/256) = 64*12 = 768 blocks
  k_gemm_qkv<<<768, 256, 0, stream>>>(xb, wqb, qkv_b, qb, kb, vtb);

  // Attention: B*H*(T/128) = 512 blocks
  k_attn<<<512, 256, 0, stream>>>(qb, kb, vtb, lambda_raw, ob);

  // Output projection: (4096/64) * (1024/256) = 64*4 = 256 blocks
  k_gemm_proj<<<256, 256, 0, stream>>>(ob, wpb, proj_b, (float*)d_out);
}
